// AttentionBlock_46420006535382
// MI455X (gfx1250) — matrix operand order repaired and checked by reading, not compiled, not run on hardware
//
#include <hip/hip_runtime.h>

// Problem constants (B,C,H,W = 32,512,32,32; N = H*W = 1024; GROUPS = 8)
#define BATCH 32
#define CCH   512
#define NSP   1024
#define LDK   40            // padded LDS k-stride in halfs (80B: keeps 16B alignment)
#define BUFH  (128 * LDK)   // halfs per LDS tile buffer

typedef __attribute__((ext_vector_type(16))) _Float16 v16h;
typedef __attribute__((ext_vector_type(8)))  _Float16 v8h;
typedef __attribute__((ext_vector_type(8)))  float    v8f;

// ---------------- WMMA fragment loaders (CDNA5 16x16x32 f16 layout) --------
// A (16x32, MxK): lane<16 -> m=lane, k = 0..7 & 16..23 ; lane>=16 -> k = 8..15 & 24..31
static __device__ inline v16h frag_a(const _Float16* t, int lane) {
  int m  = lane & 15;
  int k0 = (lane < 16) ? 0 : 8;
  v8h lo = *(const v8h*)(t + m * LDK + k0);
  v8h hi = *(const v8h*)(t + m * LDK + k0 + 16);
  v16h r;
#pragma unroll
  for (int i = 0; i < 8; ++i) { r[i] = lo[i]; r[i + 8] = hi[i]; }
  return r;
}
// B (32x16, KxN), tile stored [n][k]: lane<16 -> n=lane, k = 0..7 & 16..23 ; lane>=16 -> k = 8..15 & 24..31
static __device__ inline v16h frag_b(const _Float16* t, int lane) {
  int n  = lane & 15;
  int k0 = (lane < 16) ? 0 : 8;
  v8h lo = *(const v8h*)(t + n * LDK + k0);
  v8h hi = *(const v8h*)(t + n * LDK + k0 + 16);
  v16h r;
#pragma unroll
  for (int i = 0; i < 8; ++i) { r[i] = lo[i]; r[i + 8] = hi[i]; }
  return r;
}

// async global->LDS 16B DMA (ASYNCcnt-tracked); lds = 32-bit LDS byte offset
static __device__ inline void async_cp16(unsigned lds, const _Float16* g) {
  asm volatile("global_load_async_to_lds_b128 %0, %1, off"
               :: "v"(lds), "v"((unsigned long long)(size_t)g)
               : "memory");
}
static __device__ inline void wait_async0() {
  asm volatile("s_wait_asynccnt 0x0" ::: "memory");
}
static __device__ inline unsigned lds_off(const _Float16* p) {
  return (unsigned)(size_t)p;   // generic->LDS aperture: low 32 bits are LDS addr
}

// ---------------- shared GEMM core: 128x128 block tile, 8 waves ------------
// Double-buffered LDS; A (and B when !TB) staged with async DMA; TB's B is
// global [K][N] transposed through VGPRs with stores deferred one iteration.
template <bool TB>
static __device__ inline void gemm_core(const _Float16* __restrict__ Ag, int lda, int m0,
                                        const _Float16* __restrict__ Bg, int ldb, int n0,
                                        int ksteps, _Float16* sA, _Float16* sB,
                                        v8f acc[2][4]) {
  const int tid  = threadIdx.x;
  const int lane = tid & 31;
  const int wav  = tid >> 5;
  const int wm   = wav & 3;
  const int wn   = wav >> 2;

  // per-thread 16B chunk coordinates (row-staged tiles: 128 rows x 32 halfs)
  const int ra = tid >> 2, ca = (tid & 3) * 8;       // chunk 0 ; chunk 1 = row+64
  // TB transpose coordinates (global tile 32 k x 128 n)
  const int kb = tid >> 4, nb = (tid & 15) * 8;      // chunk 0 ; chunk 1 = k+16

  v8h pend0{}, pend1{};  // TB pending data (held across compute)

  auto issue_rows = [&](const _Float16* G, int ld, int r0, _Float16* s, int ks) {
    async_cp16(lds_off(s + ra * LDK + ca),        G + (size_t)(r0 + ra) * ld + ks * 32 + ca);
    async_cp16(lds_off(s + (ra + 64) * LDK + ca), G + (size_t)(r0 + ra + 64) * ld + ks * 32 + ca);
  };
  auto load_tb = [&](int ks) {
    pend0 = *(const v8h*)(Bg + (size_t)(ks * 32 + kb) * ldb + n0 + nb);
    pend1 = *(const v8h*)(Bg + (size_t)(ks * 32 + kb + 16) * ldb + n0 + nb);
  };
  auto store_tb = [&](_Float16* s) {
#pragma unroll
    for (int i = 0; i < 8; ++i) s[(nb + i) * LDK + kb] = pend0[i];
#pragma unroll
    for (int i = 0; i < 8; ++i) s[(nb + i) * LDK + kb + 16] = pend1[i];
  };

  // prologue: stage tile 0 into buffer 0
  issue_rows(Ag, lda, m0, sA, 0);
  if (TB) load_tb(0); else issue_rows(Bg, ldb, n0, sB, 0);

  for (int ks = 0; ks < ksteps; ++ks) {
    const int cur = ks & 1, nxt = cur ^ 1;
    if (TB) store_tb(sB + cur * BUFH);   // finish cur tile (ds stores overlap prev compute)
    wait_async0();                       // cur tile DMA done (issued one iter ago)
    __syncthreads();                     // cur tiles visible to all waves
    if (ks + 1 < ksteps) {               // prefetch next tile into other buffer
      issue_rows(Ag, lda, m0, sA + nxt * BUFH, ks + 1);
      if (TB) load_tb(ks + 1); else issue_rows(Bg, ldb, n0, sB + nxt * BUFH, ks + 1);
    }
    const _Float16* cA = sA + cur * BUFH;
    const _Float16* cB = sB + cur * BUFH;
    v16h a0 = frag_a(cA + (wm * 32) * LDK, lane);
    v16h a1 = frag_a(cA + (wm * 32 + 16) * LDK, lane);
#pragma unroll
    for (int t = 0; t < 4; ++t) {
      v16h bf = frag_b(cB + (wn * 64 + t * 16) * LDK, lane);
      acc[0][t] = __builtin_amdgcn_wmma_f32_16x16x32_f16(false, a0, false, bf,
                                                         (short)0, acc[0][t], false, false);
      acc[1][t] = __builtin_amdgcn_wmma_f32_16x16x32_f16(false, a1, false, bf,
                                                         (short)0, acc[1][t], false, false);
    }
    __syncthreads();                     // all reads done before cur is overwritten
  }
}

// ---------------- kernels --------------------------------------------------
__global__ void __launch_bounds__(256)
cvtw_kernel(const float* __restrict__ wq, const float* __restrict__ wk,
            const float* __restrict__ wv, const float* __restrict__ wo,
            _Float16* __restrict__ out) {
  int i = blockIdx.x * 256 + threadIdx.x;      // 0 .. 4*512*512-1
  const int M = CCH * CCH;
  int sel = i / M, off = i - sel * M;
  const float* s = (sel == 0) ? wq : (sel == 1) ? wk : (sel == 2) ? wv : wo;
  out[i] = (_Float16)s[off];
}

__global__ void __launch_bounds__(256)
gn_kernel(const float* __restrict__ x, const float* __restrict__ gw,
          const float* __restrict__ gb, _Float16* __restrict__ hn) {
  const int b = blockIdx.x >> 3, g = blockIdx.x & 7;   // 8 groups, 64 ch each
  const float* xp = x + ((size_t)b * CCH + g * 64) * NSP;
  float s = 0.f, s2 = 0.f;
  for (int i = threadIdx.x; i < 64 * NSP; i += 256) {
    float v = xp[i]; s += v; s2 += v * v;
  }
#pragma unroll
  for (int o = 16; o; o >>= 1) { s += __shfl_xor(s, o, 32); s2 += __shfl_xor(s2, o, 32); }
  __shared__ float as[8], as2[8];
  int wv = threadIdx.x >> 5, ln = threadIdx.x & 31;
  if (ln == 0) { as[wv] = s; as2[wv] = s2; }
  __syncthreads();
  if (wv == 0) {
    s  = (ln < 8) ? as[ln]  : 0.f;
    s2 = (ln < 8) ? as2[ln] : 0.f;
#pragma unroll
    for (int o = 4; o; o >>= 1) { s += __shfl_xor(s, o, 32); s2 += __shfl_xor(s2, o, 32); }
    if (ln == 0) { as[0] = s; as2[0] = s2; }
  }
  __syncthreads();
  const float inv = 1.f / (64.f * (float)NSP);
  float mu   = as[0] * inv;
  float var  = as2[0] * inv - mu * mu;
  float rstd = rsqrtf(var + 1e-5f);
  _Float16* hp = hn + ((size_t)b * CCH + g * 64) * NSP;
  for (int i = threadIdx.x; i < 64 * NSP; i += 256) {
    int c = g * 64 + (i >> 10);
    hp[i] = (_Float16)(((xp[i] - mu) * rstd) * gw[c] + gb[c]);
  }
}

// QKV: out = W(512x512) x hn(512x1024) + bias, per batch, per projection.
// q/k stored transposed [B,N,C] (scale folded into q); v stored [B,C,N].
__global__ void __launch_bounds__(256)
qkv_kernel(const _Float16* __restrict__ wf, const _Float16* __restrict__ hn,
           const float* __restrict__ bq, const float* __restrict__ bk,
           const float* __restrict__ bv,
           _Float16* __restrict__ qT, _Float16* __restrict__ kT,
           _Float16* __restrict__ vv) {
  __shared__ _Float16 sA[2 * BUFH];
  __shared__ _Float16 sB[2 * BUFH];
  const int bb = blockIdx.z / 3, p = blockIdx.z % 3;
  const int n0 = blockIdx.x * 128;   // spatial
  const int m0 = blockIdx.y * 128;   // cout
  const _Float16* Ag = wf + (size_t)p * CCH * CCH;
  const _Float16* Bg = hn + (size_t)bb * CCH * NSP;
  v8f acc[2][4];
#pragma unroll
  for (int i = 0; i < 2; ++i)
#pragma unroll
    for (int j = 0; j < 4; ++j) acc[i][j] = 0.f;
  gemm_core<true>(Ag, CCH, m0, Bg, NSP, n0, CCH / 32, sA, sB, acc);

  const int lane = threadIdx.x & 31, wav = threadIdx.x >> 5;
  const int wm = wav & 3, wn = wav >> 2;
  const int nloc = lane & 15, mb = (lane < 16) ? 0 : 8;
  if (p != 2) {
    _Float16* outp = ((p == 0) ? qT : kT) + (size_t)bb * NSP * CCH;
    const float* bias = (p == 0) ? bq : bk;
    const float scl = (p == 0) ? 0.04419417382415922f : 1.0f;  // 512^-0.5
#pragma unroll
    for (int mi = 0; mi < 2; ++mi)
#pragma unroll
      for (int nt = 0; nt < 4; ++nt) {
        int mcol = m0 + wm * 32 + mi * 16 + mb;
        int nrow = n0 + wn * 64 + nt * 16 + nloc;
        v8f bv8 = *(const v8f*)(bias + mcol);
        v8h h;
#pragma unroll
        for (int i = 0; i < 8; ++i) h[i] = (_Float16)((acc[mi][nt][i] + bv8[i]) * scl);
        *(v8h*)(outp + (size_t)nrow * CCH + mcol) = h;
      }
  } else {
    _Float16* outp = vv + (size_t)bb * CCH * NSP;
#pragma unroll
    for (int mi = 0; mi < 2; ++mi)
#pragma unroll
      for (int nt = 0; nt < 4; ++nt) {
        int mcol = m0 + wm * 32 + mi * 16 + mb;
        int nrow = n0 + wn * 64 + nt * 16 + nloc;
        v8f bv8 = *(const v8f*)(bv + mcol);
#pragma unroll
        for (int i = 0; i < 8; ++i)
          outp[(size_t)(mcol + i) * NSP + nrow] = (_Float16)(acc[mi][nt][i] + bv8[i]);
      }
  }
}

// S[b][i][j] = sum_c qT[b][i][c] * kT[b][j][c]  (A = kT rows -> M=j, B = qT rows -> N=i)
__global__ void __launch_bounds__(256)
scores_kernel(const _Float16* __restrict__ kT, const _Float16* __restrict__ qT,
              _Float16* __restrict__ S) {
  __shared__ _Float16 sA[2 * BUFH];
  __shared__ _Float16 sB[2 * BUFH];
  const int bb = blockIdx.z;
  const int n0 = blockIdx.x * 128;  // i
  const int m0 = blockIdx.y * 128;  // j
  const _Float16* Ag = kT + (size_t)bb * NSP * CCH;
  const _Float16* Bg = qT + (size_t)bb * NSP * CCH;
  v8f acc[2][4];
#pragma unroll
  for (int i = 0; i < 2; ++i)
#pragma unroll
    for (int j = 0; j < 4; ++j) acc[i][j] = 0.f;
  gemm_core<false>(Ag, CCH, m0, Bg, CCH, n0, CCH / 32, sA, sB, acc);

  const int lane = threadIdx.x & 31, wav = threadIdx.x >> 5;
  const int wm = wav & 3, wn = wav >> 2;
  const int nloc = lane & 15, mb = (lane < 16) ? 0 : 8;
  _Float16* outp = S + (size_t)bb * NSP * NSP;
#pragma unroll
  for (int mi = 0; mi < 2; ++mi)
#pragma unroll
    for (int nt = 0; nt < 4; ++nt) {
      int mcol = m0 + wm * 32 + mi * 16 + mb;        // j (contiguous)
      int nrow = n0 + wn * 64 + nt * 16 + nloc;      // i
      v8h h;
#pragma unroll
      for (int i = 0; i < 8; ++i) h[i] = (_Float16)acc[mi][nt][i];
      *(v8h*)(outp + (size_t)nrow * NSP + mcol) = h;
    }
}

// in-place row softmax over 1024 f16 values; one wave per row
__global__ void __launch_bounds__(256)
softmax_kernel(_Float16* __restrict__ S) {
  const int row  = blockIdx.x * 8 + (threadIdx.x >> 5);
  const int lane = threadIdx.x & 31;
  _Float16* p = S + (size_t)row * NSP + lane * 32;
  float v[32];
#pragma unroll
  for (int t = 0; t < 4; ++t) {
    v8h d = *(const v8h*)(p + t * 8);
#pragma unroll
    for (int i = 0; i < 8; ++i) v[t * 8 + i] = (float)d[i];
  }
  float mx = -3.0e38f;
#pragma unroll
  for (int i = 0; i < 32; ++i) mx = fmaxf(mx, v[i]);
#pragma unroll
  for (int o = 16; o; o >>= 1) mx = fmaxf(mx, __shfl_xor(mx, o, 32));
  float sum = 0.f;
#pragma unroll
  for (int i = 0; i < 32; ++i) { v[i] = __expf(v[i] - mx); sum += v[i]; }
#pragma unroll
  for (int o = 16; o; o >>= 1) sum += __shfl_xor(sum, o, 32);
  float inv = 1.f / sum;
#pragma unroll
  for (int t = 0; t < 4; ++t) {
    v8h d;
#pragma unroll
    for (int i = 0; i < 8; ++i) d[i] = (_Float16)(v[t * 8 + i] * inv);
    *(v8h*)(p + t * 8) = d;
  }
}

// oT[b][i][c] = sum_j v[b][c][j] * P[b][i][j]   (A = v rows -> M=c, B = P rows -> N=i)
__global__ void __launch_bounds__(256)
pv_kernel(const _Float16* __restrict__ vv, const _Float16* __restrict__ P,
          _Float16* __restrict__ oT) {
  __shared__ _Float16 sA[2 * BUFH];
  __shared__ _Float16 sB[2 * BUFH];
  const int bb = blockIdx.z;
  const int n0 = blockIdx.x * 128;  // i
  const int m0 = blockIdx.y * 128;  // c
  const _Float16* Ag = vv + (size_t)bb * CCH * NSP;
  const _Float16* Bg = P + (size_t)bb * NSP * NSP;
  v8f acc[2][4];
#pragma unroll
  for (int i = 0; i < 2; ++i)
#pragma unroll
    for (int j = 0; j < 4; ++j) acc[i][j] = 0.f;
  gemm_core<false>(Ag, NSP, m0, Bg, NSP, n0, NSP / 32, sA, sB, acc);

  const int lane = threadIdx.x & 31, wav = threadIdx.x >> 5;
  const int wm = wav & 3, wn = wav >> 2;
  const int nloc = lane & 15, mb = (lane < 16) ? 0 : 8;
  _Float16* outp = oT + (size_t)bb * NSP * CCH;
#pragma unroll
  for (int mi = 0; mi < 2; ++mi)
#pragma unroll
    for (int nt = 0; nt < 4; ++nt) {
      int mcol = m0 + wm * 32 + mi * 16 + mb;        // c (contiguous)
      int nrow = n0 + wn * 64 + nt * 16 + nloc;      // i
      v8h h;
#pragma unroll
      for (int i = 0; i < 8; ++i) h[i] = (_Float16)acc[mi][nt][i];
      *(v8h*)(outp + (size_t)nrow * CCH + mcol) = h;
    }
}

// out[b][cout][i] = x + sum_c wo[cout][c]*oT[b][i][c] + bo
// (A = oT rows -> M=i, B = wo rows -> N=cout)
__global__ void __launch_bounds__(256)
proj_kernel(const _Float16* __restrict__ oT, const _Float16* __restrict__ woF,
            const float* __restrict__ xin, const float* __restrict__ bo,
            float* __restrict__ out) {
  __shared__ _Float16 sA[2 * BUFH];
  __shared__ _Float16 sB[2 * BUFH];
  const int bb = blockIdx.z;
  const int n0 = blockIdx.x * 128;  // cout
  const int m0 = blockIdx.y * 128;  // i
  const _Float16* Ag = oT + (size_t)bb * NSP * CCH;
  v8f acc[2][4];
#pragma unroll
  for (int i = 0; i < 2; ++i)
#pragma unroll
    for (int j = 0; j < 4; ++j) acc[i][j] = 0.f;
  gemm_core<false>(Ag, CCH, m0, woF, CCH, n0, CCH / 32, sA, sB, acc);

  const int lane = threadIdx.x & 31, wav = threadIdx.x >> 5;
  const int wm = wav & 3, wn = wav >> 2;
  const int nloc = lane & 15, mb = (lane < 16) ? 0 : 8;
#pragma unroll
  for (int mi = 0; mi < 2; ++mi)
#pragma unroll
    for (int nt = 0; nt < 4; ++nt) {
      int irow = m0 + wm * 32 + mi * 16 + mb;        // i (8 contiguous)
      int cc   = n0 + wn * 64 + nt * 16 + nloc;      // cout
      size_t off = ((size_t)bb * CCH + cc) * NSP + irow;
      v8f xv = *(const v8f*)(xin + off);
      float bc = bo[cc];
      v8f r;
#pragma unroll
      for (int i = 0; i < 8; ++i) r[i] = acc[mi][nt][i] + bc + xv[i];
      *(v8f*)(out + off) = r;
    }
}

// ---------------- launch ---------------------------------------------------
extern "C" void kernel_launch(void* const* d_in, const int* in_sizes, int n_in,
                              void* d_out, int out_size, void* d_ws, size_t ws_size,
                              hipStream_t stream) {
  (void)in_sizes; (void)n_in; (void)out_size; (void)ws_size;
  const float* x    = (const float*)d_in[0];
  const float* gn_w = (const float*)d_in[1];
  const float* gn_b = (const float*)d_in[2];
  const float* wq   = (const float*)d_in[3];
  const float* bq   = (const float*)d_in[4];
  const float* wk   = (const float*)d_in[5];
  const float* bk   = (const float*)d_in[6];
  const float* wv   = (const float*)d_in[7];
  const float* bv   = (const float*)d_in[8];
  const float* wo   = (const float*)d_in[9];
  const float* bo   = (const float*)d_in[10];
  float* out = (float*)d_out;

  // workspace carve-up (halfs)
  _Float16* wf = (_Float16*)d_ws;                           // 4*512*512
  _Float16* hn = wf + (size_t)4 * CCH * CCH;                // [B,C,N]
  _Float16* qT = hn + (size_t)BATCH * CCH * NSP;            // [B,N,C]
  _Float16* kT = qT + (size_t)BATCH * NSP * CCH;            // [B,N,C]
  _Float16* vv = kT + (size_t)BATCH * NSP * CCH;            // [B,C,N]
  _Float16* Sm = vv + (size_t)BATCH * CCH * NSP;            // [B,N,N]
  _Float16* oT = Sm + (size_t)BATCH * NSP * NSP;            // [B,N,C]

  cvtw_kernel<<<(4 * CCH * CCH) / 256, 256, 0, stream>>>(wq, wk, wv, wo, wf);
  gn_kernel<<<BATCH * 8, 256, 0, stream>>>(x, gn_w, gn_b, hn);
  qkv_kernel<<<dim3(NSP / 128, CCH / 128, BATCH * 3), 256, 0, stream>>>(
      wf, hn, bq, bk, bv, qT, kT, vv);
  scores_kernel<<<dim3(NSP / 128, NSP / 128, BATCH), 256, 0, stream>>>(kT, qT, Sm);
  softmax_kernel<<<(BATCH * NSP) / 8, 256, 0, stream>>>(Sm);
  pv_kernel<<<dim3(NSP / 128, CCH / 128, BATCH), 256, 0, stream>>>(vv, Sm, oT);
  proj_kernel<<<dim3(CCH / 128, NSP / 128, BATCH), 256, 0, stream>>>(
      oT, wf + (size_t)3 * CCH * CCH, x, bo, out);
}
